// HyperAttnLayer_2576980378157
// MI455X (gfx1250) — matrix operand order repaired and checked by reading, compiled, not run on hardware
//
#include <hip/hip_runtime.h>
#include <hip/hip_bf16.h>

typedef __attribute__((ext_vector_type(16))) _Float16 v16h;
typedef __attribute__((ext_vector_type(8)))  _Float16 v8h;
typedef __attribute__((ext_vector_type(8)))  float    v8f;

namespace {
constexpr int N_NODES = 50000;
constexpr int N_EDGES = 10000;
constexpr int NNZ     = 400000;
constexpr int QDIM    = 128;
constexpr int VDIM    = 256;
constexpr int EDIM    = 256;
constexpr int KDIM    = 256;   // all GEMMs have K = 256
constexpr float NEG_SLOPE = 0.01f;
constexpr float SCALE = 0.08838834764831845f; // 1/sqrt(128)

__host__ __device__ inline int cdiv(int a, int b) { return (a + b - 1) / b; }
} // namespace

// ---------------- elementwise helpers ----------------

__global__ void k_f32_to_f16(const float* __restrict__ in, _Float16* __restrict__ out, int n) {
    int i = blockIdx.x * blockDim.x + threadIdx.x;
    if (i < n) out[i] = (_Float16)in[i];
}

__global__ void k_fill(float* __restrict__ p, float v, int n) {
    int i = blockIdx.x * blockDim.x + threadIdx.x;
    if (i < n) p[i] = v;
}

__global__ void k_relu_cvt(float* __restrict__ x, _Float16* __restrict__ h, int n) {
    int i = blockIdx.x * blockDim.x + threadIdx.x;
    if (i >= n) return;
    float v = fmaxf(x[i], 0.0f);
    x[i] = v;
    if (h) h[i] = (_Float16)v;
}

// ---------------- WMMA GEMM: C[M,N] = A[M,K] * W[N,K]^T + bias ----------------
// Register-blocked: one wave32 computes a 16x64 strip of C (4 accumulators).
// Per K-step (32): load the A fragment ONCE, 4 B fragments, issue 4 WMMAs.
//
// A fragment (16x32 f16): lane L holds row m=L&15; half-elements:
//   elems 0..7  -> K = k0 + (L>=16 ? 16 : 0) + 0..7
//   elems 8..15 -> K = k0 + (L>=16 ? 16 : 0) + 8..15
// B fragment (32x16 f16, B[k][n] = W[n][k]): lane L holds n=L&15;
//   elems 0..15 -> K = k0 + (L>=16 ? 16 : 0) + 0..15
// C/D layout: lane L -> n = L&15 ; VGPR r -> m = r + (L>=16 ? 8 : 0)

__device__ __forceinline__ v16h ldfrag(const _Float16* __restrict__ p, int gap_halves) {
    v8h lo = *(const v8h*)(p);
    v8h hi = *(const v8h*)(p + gap_halves);
    v16h r;
#pragma unroll
    for (int i = 0; i < 8; ++i) { r[i] = lo[i]; r[8 + i] = hi[i]; }
    return r;
}

__global__ void k_gemm_wmma(const _Float16* __restrict__ A,
                            const _Float16* __restrict__ W,
                            const float* __restrict__ bias,
                            float* __restrict__ outF,
                            _Float16* __restrict__ outH,
                            int M, int N, int K)
{
    const int lane = threadIdx.x & 31;
    const int wave = threadIdx.x >> 5;
    const int ngrp = N >> 6;                        // 64-wide N groups
    const int tilesTot = (M >> 4) * ngrp;
    const int tile = blockIdx.x * (blockDim.x >> 5) + wave;
    if (tile >= tilesTot) return;                   // wave-uniform exit (EXEC all-1s past here)

    const int tm = tile / ngrp;
    const int n0 = (tile - tm * ngrp) * 64;
    const int r16 = lane & 15;
    const int hi  = lane >> 4;

    // per-lane base pointers (fragment element selection folded in)
    const _Float16* Ap  = A + (size_t)(tm * 16 + r16) * K + hi * 16;
    const _Float16* Wp0 = W + (size_t)(n0 + r16) * K + hi * 16;
    const size_t wstride = (size_t)16 * K;          // 16 rows of W per N-tile

    v8f c0, c1, c2, c3;
    {
        const float b0 = bias[n0 + r16];
        const float b1 = bias[n0 + 16 + r16];
        const float b2 = bias[n0 + 32 + r16];
        const float b3 = bias[n0 + 48 + r16];
#pragma unroll
        for (int r = 0; r < 8; ++r) { c0[r] = b0; c1[r] = b1; c2[r] = b2; c3[r] = b3; }
    }

    for (int k0 = 0; k0 < K; k0 += 32) {
        v16h a  = ldfrag(Ap + k0, 8);               // A fragment reused 4x
        v16h b0 = ldfrag(Wp0 + k0, 8);
        v16h b1 = ldfrag(Wp0 + wstride + k0, 8);
        v16h b2 = ldfrag(Wp0 + 2 * wstride + k0, 8);
        v16h b3 = ldfrag(Wp0 + 3 * wstride + k0, 8);
        c0 = __builtin_amdgcn_wmma_f32_16x16x32_f16(false, a, false, b0, (short)0, c0, false, false);
        c1 = __builtin_amdgcn_wmma_f32_16x16x32_f16(false, a, false, b1, (short)0, c1, false, false);
        c2 = __builtin_amdgcn_wmma_f32_16x16x32_f16(false, a, false, b2, (short)0, c2, false, false);
        c3 = __builtin_amdgcn_wmma_f32_16x16x32_f16(false, a, false, b3, (short)0, c3, false, false);
    }

    const int mbase = tm * 16 + hi * 8;
    v8f acc[4] = {c0, c1, c2, c3};
#pragma unroll
    for (int j = 0; j < 4; ++j) {
        const int n = n0 + j * 16 + r16;
#pragma unroll
        for (int r = 0; r < 8; ++r) {
            float v = acc[j][r];
            if (outF) outF[(size_t)(mbase + r) * N + n] = v;
            if (outH) outH[(size_t)(mbase + r) * N + n] = (_Float16)v;
        }
    }
}

// ---------------- attention score / softmax / scatter ----------------

__device__ __forceinline__ void atomicMaxF(float* addr, float v) {
    if (v >= 0.0f) atomicMax((int*)addr, __float_as_int(v));
    else           atomicMin((unsigned int*)addr, __float_as_uint(v));
}

// one wave per nnz; QDIM = 128 -> one float4 per lane
__global__ void k_score(const float* __restrict__ Kf, const float* __restrict__ Qf,
                        const int* __restrict__ kidx, const int* __restrict__ qidx,
                        float* __restrict__ score, float* __restrict__ segmax, int nnz)
{
    const int wid = blockIdx.x * (blockDim.x >> 5) + (threadIdx.x >> 5);
    if (wid >= nnz) return;
    const int lane = threadIdx.x & 31;
    const float4* kp = (const float4*)(Kf + (size_t)kidx[wid] * QDIM);
    const float4* qp = (const float4*)(Qf + (size_t)qidx[wid] * QDIM);
    float4 kv = kp[lane];
    float4 qv = qp[lane];
    float s = kv.x * qv.x + kv.y * qv.y + kv.z * qv.z + kv.w * qv.w;
#pragma unroll
    for (int off = 16; off >= 1; off >>= 1) s += __shfl_xor(s, off);
    if (lane == 0) {
        s = (s >= 0.0f ? s : NEG_SLOPE * s) * SCALE;
        score[wid] = s;
        atomicMaxF(&segmax[qidx[wid]], s);
    }
}

__global__ void k_expsum(float* __restrict__ score, const float* __restrict__ segmax,
                         float* __restrict__ segsum, const int* __restrict__ seg, int nnz)
{
    int i = blockIdx.x * blockDim.x + threadIdx.x;
    if (i >= nnz) return;
    int s = seg[i];
    float e = __expf(score[i] - segmax[s]);
    score[i] = e;
    atomicAdd(&segsum[s], e);
}

// one wave per nnz; Out[seg] += (e/sum) * V[vidx]  over vd dims
__global__ void k_scatter(const float* __restrict__ score, const float* __restrict__ segsum,
                          const int* __restrict__ seg, const int* __restrict__ vidx,
                          const float* __restrict__ V, float* __restrict__ Out,
                          int nnz, int vd)
{
    const int wid = blockIdx.x * (blockDim.x >> 5) + (threadIdx.x >> 5);
    if (wid >= nnz) return;
    const int lane = threadIdx.x & 31;
    const int s = seg[wid];
    const float w = score[wid] / fmaxf(segsum[s], 1e-20f);
    const float* src = V + (size_t)vidx[wid] * vd;
    float* dst = Out + (size_t)s * vd;
    for (int d = lane; d < vd; d += 32) atomicAdd(&dst[d], w * src[d]);
}

// ---------------- host-side launch ----------------

extern "C" void kernel_launch(void* const* d_in, const int* in_sizes, int n_in,
                              void* d_out, int out_size, void* d_ws, size_t ws_size,
                              hipStream_t stream) {
    const float* vfeat = (const float*)d_in[0];
    const float* efeat = (const float*)d_in[1];
    const int* node_idx = (const int*)d_in[2];
    const int* edge_idx = (const int*)d_in[3];
    const float* W_vtx = (const float*)d_in[4];  const float* b_vtx = (const float*)d_in[5];
    const float* W_qe  = (const float*)d_in[6];  const float* b_qe  = (const float*)d_in[7];
    const float* W_kv  = (const float*)d_in[8];  const float* b_kv  = (const float*)d_in[9];
    const float* W_vv  = (const float*)d_in[10]; const float* b_vv  = (const float*)d_in[11];
    const float* W_qv  = (const float*)d_in[12]; const float* b_qv  = (const float*)d_in[13];
    const float* W_ke  = (const float*)d_in[14]; const float* b_ke  = (const float*)d_in[15];
    const float* W_ve  = (const float*)d_in[16]; const float* b_ve  = (const float*)d_in[17];

    float* out_v = (float*)d_out;                          // [N_NODES, VDIM]
    float* out_e = (float*)d_out + (size_t)N_NODES * VDIM; // [N_EDGES, EDIM]

    // ---- carve workspace (256B aligned) ----
    char* wsp = (char*)d_ws;
    auto carve = [&](size_t bytes) {
        void* p = (void*)wsp;
        wsp += (bytes + 255) & ~(size_t)255;
        return p;
    };
    _Float16* vfeat_h = (_Float16*)carve((size_t)N_NODES * KDIM * 2);
    _Float16* efeat_h = (_Float16*)carve((size_t)N_EDGES * KDIM * 2);
    _Float16* featv_h = (_Float16*)carve((size_t)N_NODES * VDIM * 2);
    _Float16* feate_h = (_Float16*)carve((size_t)N_EDGES * EDIM * 2);
    _Float16* Wvtx_h = (_Float16*)carve((size_t)VDIM * KDIM * 2);
    _Float16* Wqe_h  = (_Float16*)carve((size_t)QDIM * KDIM * 2);
    _Float16* Wkv_h  = (_Float16*)carve((size_t)QDIM * KDIM * 2);
    _Float16* Wvv_h  = (_Float16*)carve((size_t)EDIM * KDIM * 2);
    _Float16* Wqv_h  = (_Float16*)carve((size_t)QDIM * KDIM * 2);
    _Float16* Wke_h  = (_Float16*)carve((size_t)QDIM * KDIM * 2);
    _Float16* Wve_h  = (_Float16*)carve((size_t)VDIM * KDIM * 2);
    float* k1 = (float*)carve((size_t)N_NODES * QDIM * 4);
    float* v1 = (float*)carve((size_t)N_NODES * EDIM * 4);
    float* q1 = (float*)carve((size_t)N_EDGES * QDIM * 4);
    float* k2 = (float*)carve((size_t)N_EDGES * QDIM * 4);
    float* v2 = (float*)carve((size_t)N_EDGES * VDIM * 4);
    float* q2 = (float*)carve((size_t)N_NODES * QDIM * 4);
    float* score = (float*)carve((size_t)NNZ * 4);
    float* max1 = (float*)carve((size_t)N_EDGES * 4);
    float* sum1 = (float*)carve((size_t)N_EDGES * 4);
    float* max2 = (float*)carve((size_t)N_NODES * 4);
    float* sum2 = (float*)carve((size_t)N_NODES * 4);

    const int T = 256;
    auto cvt = [&](const float* src, _Float16* dst, int n) {
        k_f32_to_f16<<<cdiv(n, T), T, 0, stream>>>(src, dst, n);
    };
    auto fill = [&](float* p, float v, int n) {
        k_fill<<<cdiv(n, T), T, 0, stream>>>(p, v, n);
    };
    auto gemm = [&](const _Float16* A, const _Float16* W, const float* bias,
                    float* outF, _Float16* outH, int M, int N) {
        int tiles = (M >> 4) * (N >> 6);            // 16-row x 64-col strips per wave
        k_gemm_wmma<<<cdiv(tiles, T / 32), T, 0, stream>>>(A, W, bias, outF, outH, M, N, KDIM);
    };

    // ---- stage 0: conversions + inits ----
    cvt(vfeat, vfeat_h, N_NODES * KDIM);
    cvt(efeat, efeat_h, N_EDGES * KDIM);
    cvt(W_vtx, Wvtx_h, VDIM * KDIM);
    cvt(W_qe,  Wqe_h,  QDIM * KDIM);
    cvt(W_kv,  Wkv_h,  QDIM * KDIM);
    cvt(W_vv,  Wvv_h,  EDIM * KDIM);
    cvt(W_qv,  Wqv_h,  QDIM * KDIM);
    cvt(W_ke,  Wke_h,  QDIM * KDIM);
    cvt(W_ve,  Wve_h,  VDIM * KDIM);
    fill(out_v, 0.0f, N_NODES * VDIM);
    fill(out_e, 0.0f, N_EDGES * EDIM);
    fill(max1, -__builtin_huge_valf(), N_EDGES);
    fill(sum1, 0.0f, N_EDGES);
    fill(max2, -__builtin_huge_valf(), N_NODES);
    fill(sum2, 0.0f, N_NODES);

    // ---- pass 1: nodes -> hyperedges ----
    gemm(vfeat_h, Wvtx_h, b_vtx, nullptr, featv_h, N_NODES, VDIM);   // feat_v (f16 only)
    gemm(efeat_h, Wqe_h,  b_qe,  q1, nullptr, N_EDGES, QDIM);        // q1
    gemm(featv_h, Wkv_h,  b_kv,  k1, nullptr, N_NODES, QDIM);        // k1
    gemm(featv_h, Wvv_h,  b_vv,  v1, nullptr, N_NODES, EDIM);        // v1
    gemm(featv_h, Wqv_h,  b_qv,  q2, nullptr, N_NODES, QDIM);        // q2 (for pass 2)

    k_score<<<cdiv(NNZ, T / 32), T, 0, stream>>>(k1, q1, node_idx, edge_idx, score, max1, NNZ);
    k_expsum<<<cdiv(NNZ, T), T, 0, stream>>>(score, max1, sum1, edge_idx, NNZ);
    k_scatter<<<cdiv(NNZ, T / 32), T, 0, stream>>>(score, sum1, edge_idx, node_idx, v1, out_e, NNZ, EDIM);
    k_relu_cvt<<<cdiv(N_EDGES * EDIM, T), T, 0, stream>>>(out_e, feate_h, N_EDGES * EDIM);

    // ---- pass 2: hyperedges -> nodes ----
    gemm(feate_h, Wke_h, b_ke, k2, nullptr, N_EDGES, QDIM);          // k2
    gemm(feate_h, Wve_h, b_ve, v2, nullptr, N_EDGES, VDIM);          // v2

    k_score<<<cdiv(NNZ, T / 32), T, 0, stream>>>(k2, q2, edge_idx, node_idx, score, max2, NNZ);
    k_expsum<<<cdiv(NNZ, T), T, 0, stream>>>(score, max2, sum2, node_idx, NNZ);
    k_scatter<<<cdiv(NNZ, T / 32), T, 0, stream>>>(score, sum2, node_idx, edge_idx, v2, out_v, NNZ, VDIM);
    k_relu_cvt<<<cdiv(N_NODES * VDIM, T), T, 0, stream>>>(out_v, nullptr, N_NODES * VDIM);
}
